// TtCausalSelfAttention_42743514530654
// MI455X (gfx1250) — matrix operand order repaired and checked by reading, not compiled, not run on hardware
//
#include <hip/hip_runtime.h>
#include <hip/hip_bf16.h>
#include <stdint.h>

// ---- problem dims (match reference) ----
#define BB 2
#define TT 2048
#define CC 1024
#define HH 16
#define HD 64
#define MROWS (BB * TT)      // 4096 token rows
#define N_QKV (3 * CC)       // 3072

typedef __attribute__((ext_vector_type(16))) __bf16 v16bf;
typedef __attribute__((ext_vector_type(8)))  float  v8f;

union FragAB { uint4 u4[2]; unsigned u[8]; v16bf v; };  // 32B: one A or B fragment
union FragC  { float f[8]; v8f v; };                    // 32B: one C/D fragment

__device__ __forceinline__ v8f wmma_bf16(const v16bf& a, const v16bf& b, const v8f& c) {
  return __builtin_amdgcn_wmma_f32_16x16x32_bf16(false, a, false, b, (short)0, c, false, false);
}

__device__ __forceinline__ void load_frag(const __bf16* p0, const __bf16* p1, FragAB& f) {
  f.u4[0] = *reinterpret_cast<const uint4*>(p0);
  f.u4[1] = *reinterpret_cast<const uint4*>(p1);
}

// XOR lane shuffle via ds_swizzle (BITMASK_PERM: offset = xor<<10 | and 0x1f), wave32 native.
#define SWZ_XOR(v, M) \
  __int_as_float(__builtin_amdgcn_ds_swizzle(__float_as_int(v), (((M) << 10) | 0x1f)))

__device__ __forceinline__ float redmax16(float v) {
  v = fmaxf(v, SWZ_XOR(v, 1)); v = fmaxf(v, SWZ_XOR(v, 2));
  v = fmaxf(v, SWZ_XOR(v, 4)); v = fmaxf(v, SWZ_XOR(v, 8));
  return v;
}
__device__ __forceinline__ float redsum16(float v) {
  v += SWZ_XOR(v, 1); v += SWZ_XOR(v, 2); v += SWZ_XOR(v, 4); v += SWZ_XOR(v, 8);
  return v;
}

// ---------------- kernel 0: fp32 -> bf16 ----------------
__global__ void cvt_f32_bf16(const float* __restrict__ src, __bf16* __restrict__ dst, int n) {
  for (int i = blockIdx.x * blockDim.x + threadIdx.x; i < n; i += gridDim.x * blockDim.x)
    dst[i] = (__bf16)src[i];
}

// ---------------- kernel 1: QKV GEMM (32x64 tile per wave) ----------------
// qkv = x @ w_attn^T + b_attn ; scatter into Q[B,H,T,64], K[B,H,T,64], Vt[B,H,64,T] (bf16)
__global__ __launch_bounds__(128) void qkv_gemm(
    const __bf16* __restrict__ X, const __bf16* __restrict__ W,
    const float* __restrict__ bias,
    __bf16* __restrict__ Q, __bf16* __restrict__ K, __bf16* __restrict__ Vt) {
  const int lane = threadIdx.x & 31;
  const int gw   = blockIdx.x * 4 + (threadIdx.x >> 5);
  const int NGRP = N_QKV / 64;                  // 48 column groups of 64
  const int mt = gw / NGRP, ng = gw % NGRP;     // mt over 128 tiles of 32 rows
  const int l16 = lane & 15, hi = (lane >> 4) & 1;

  size_t abase[2];
#pragma unroll
  for (int mi = 0; mi < 2; ++mi)
    abase[mi] = (size_t)(mt * 32 + mi * 16 + l16) * CC + (hi ? 8 : 0);
  const int nbase = ng * 64;
  size_t bbase[4];
#pragma unroll
  for (int nt = 0; nt < 4; ++nt)
    bbase[nt] = (size_t)(nbase + nt * 16 + l16) * CC + (hi ? 8 : 0);

  FragC acc[2][4];
#pragma unroll
  for (int mi = 0; mi < 2; ++mi)
#pragma unroll
    for (int nt = 0; nt < 4; ++nt)
#pragma unroll
      for (int r = 0; r < 8; ++r) acc[mi][nt].f[r] = 0.f;

  for (int k = 0; k < CC; k += 32) {
    FragAB a[2];
#pragma unroll
    for (int mi = 0; mi < 2; ++mi)
      load_frag(X + abase[mi] + k, X + abase[mi] + k + 16, a[mi]);
#pragma unroll
    for (int nt = 0; nt < 4; ++nt) {
      FragAB b;
      const __bf16* bp = W + bbase[nt] + k;
      load_frag(bp, bp + 16, b);
      acc[0][nt].v = wmma_bf16(a[0].v, b.v, acc[0][nt].v);
      acc[1][nt].v = wmma_bf16(a[1].v, b.v, acc[1][nt].v);
    }
  }

#pragma unroll
  for (int nt = 0; nt < 4; ++nt) {
    const int n = nbase + nt * 16 + l16;
    const float bn = bias[n];
#pragma unroll
    for (int mi = 0; mi < 2; ++mi)
#pragma unroll
      for (int r = 0; r < 8; ++r) {
        const int m = mt * 32 + mi * 16 + r + hi * 8;
        const int bidx = m / TT, t = m % TT;
        const float val = acc[mi][nt].f[r] + bn;
        if (n < CC) {
          const int h = n >> 6, d = n & 63;
          Q[(((size_t)(bidx * HH + h) * TT) + t) * HD + d] = (__bf16)val;
        } else if (n < 2 * CC) {
          const int n2 = n - CC, h = n2 >> 6, d = n2 & 63;
          K[(((size_t)(bidx * HH + h) * TT) + t) * HD + d] = (__bf16)val;
        } else {
          const int n2 = n - 2 * CC, h = n2 >> 6, d = n2 & 63;
          Vt[(((size_t)(bidx * HH + h) * HD) + d) * TT + t] = (__bf16)val;
        }
      }
  }
}

// ---------------- kernel 2: causal flash attention (32 q-rows per wave) ----------------
__global__ __launch_bounds__(128) void attn(
    const __bf16* __restrict__ Q, const __bf16* __restrict__ K,
    const __bf16* __restrict__ Vt, __bf16* __restrict__ Y) {
  __shared__ __bf16 lds[4 * 2 * 16 * 32];         // 2KB per wave: two P tiles
  const int lane = threadIdx.x & 31;
  const int wslot = threadIdx.x >> 5;
  const int gw   = blockIdx.x * 4 + wslot;
  const int QB = TT / 32;                         // 64 q-blocks per (b,h)
  const int bh = gw / QB, qblk = gw % QB;
  const int b = bh / HH, h = bh % HH;
  const int qbase = qblk * 32;
  const int l16 = lane & 15, hi = (lane >> 4) & 1;

  const __bf16* Qp = Q + (size_t)bh * TT * HD;    // [T,64]
  const __bf16* Kp = K + (size_t)bh * TT * HD;    // [T,64]
  const __bf16* Vp = Vt + (size_t)bh * HD * TT;   // [64,T]
  __bf16* my = lds + wslot * (2 * 16 * 32);

  // Q A-fragments: [mi][ks] with d-chunks 0..31 / 32..63
  FragAB qf[2][2];
#pragma unroll
  for (int mi = 0; mi < 2; ++mi) {
    const __bf16* qr = Qp + (size_t)(qbase + mi * 16 + l16) * HD + (hi ? 8 : 0);
    load_frag(qr + 0, qr + 16, qf[mi][0]);
    load_frag(qr + 32, qr + 48, qf[mi][1]);
  }

  FragC acc[2][4];
  float mrow[2][8], lrow[2][8];
#pragma unroll
  for (int mi = 0; mi < 2; ++mi) {
#pragma unroll
    for (int nt = 0; nt < 4; ++nt)
#pragma unroll
      for (int r = 0; r < 8; ++r) acc[mi][nt].f[r] = 0.f;
#pragma unroll
    for (int r = 0; r < 8; ++r) { mrow[mi][r] = -3.0e38f; lrow[mi][r] = 0.f; }
  }

  const float scale = 0.125f;                     // 1/sqrt(64)
  const int kend = qbase + 32;

  for (int kb = 0; kb < kend; kb += 32) {
    // S = Q K^T : K-fragments shared across both M-tiles
    FragC S[2][2];
#pragma unroll
    for (int mi = 0; mi < 2; ++mi)
#pragma unroll
      for (int ns = 0; ns < 2; ++ns)
#pragma unroll
        for (int r = 0; r < 8; ++r) S[mi][ns].f[r] = 0.f;
#pragma unroll
    for (int ns = 0; ns < 2; ++ns) {
      const __bf16* kr = Kp + (size_t)(kb + ns * 16 + l16) * HD + (hi ? 8 : 0);
      FragAB k0, k1;
      load_frag(kr, kr + 16, k0);
      load_frag(kr + 32, kr + 48, k1);
#pragma unroll
      for (int mi = 0; mi < 2; ++mi) {
        S[mi][ns].v = wmma_bf16(qf[mi][0].v, k0.v, S[mi][ns].v);
        S[mi][ns].v = wmma_bf16(qf[mi][1].v, k1.v, S[mi][ns].v);
      }
    }
    // per M-tile: mask, online softmax, stash P into LDS
    float alpha[2][8];
#pragma unroll
    for (int mi = 0; mi < 2; ++mi) {
#pragma unroll
      for (int r = 0; r < 8; ++r) {
        const int qrow = qbase + mi * 16 + r + hi * 8;
        float s0 = S[mi][0].f[r] * scale; if (kb + l16 > qrow)      s0 = -3.0e38f;
        float s1 = S[mi][1].f[r] * scale; if (kb + 16 + l16 > qrow) s1 = -3.0e38f;
        S[mi][0].f[r] = s0; S[mi][1].f[r] = s1;
        const float bm = redmax16(fmaxf(s0, s1));
        const float mn = fmaxf(mrow[mi][r], bm);
        alpha[mi][r] = __expf(mrow[mi][r] - mn);
        mrow[mi][r] = mn;
      }
#pragma unroll
      for (int r = 0; r < 8; ++r) {
        const int ri = r + hi * 8;
        const float p0 = __expf(S[mi][0].f[r] - mrow[mi][r]);
        const float p1 = __expf(S[mi][1].f[r] - mrow[mi][r]);
        my[mi * 512 + ri * 32 + l16]      = (__bf16)p0;
        my[mi * 512 + ri * 32 + 16 + l16] = (__bf16)p1;
        lrow[mi][r] = lrow[mi][r] * alpha[mi][r] + redsum16(p0 + p1);
      }
    }
    asm volatile("s_wait_dscnt 0" ::: "memory");  // LDS stores visible before relayout read
    // P as A-fragments (16x32 bf16 each)
    FragAB pf[2];
#pragma unroll
    for (int mi = 0; mi < 2; ++mi) {
      const __bf16* pr = my + mi * 512 + l16 * 32 + (hi ? 8 : 0);
      load_frag(pr, pr + 16, pf[mi]);
    }
    // rescale accumulators, then acc += P @ V (V-fragments shared across M-tiles)
#pragma unroll
    for (int mi = 0; mi < 2; ++mi)
#pragma unroll
      for (int nt = 0; nt < 4; ++nt)
#pragma unroll
        for (int r = 0; r < 8; ++r) acc[mi][nt].f[r] *= alpha[mi][r];
#pragma unroll
    for (int nt = 0; nt < 4; ++nt) {
      const __bf16* vr = Vp + (size_t)(nt * 16 + l16) * TT + kb + (hi ? 8 : 0);
      FragAB vf;
      load_frag(vr, vr + 16, vf);
      acc[0][nt].v = wmma_bf16(pf[0].v, vf.v, acc[0][nt].v);
      acc[1][nt].v = wmma_bf16(pf[1].v, vf.v, acc[1][nt].v);
    }
  }

  // normalize and emit y (bf16, [B,T,C] with c = h*64 + d)
#pragma unroll
  for (int mi = 0; mi < 2; ++mi) {
    float inv[8];
#pragma unroll
    for (int r = 0; r < 8; ++r) inv[r] = 1.0f / lrow[mi][r];
#pragma unroll
    for (int nt = 0; nt < 4; ++nt) {
      const int d = nt * 16 + l16;
#pragma unroll
      for (int r = 0; r < 8; ++r) {
        const int t = qbase + mi * 16 + r + hi * 8;
        Y[(((size_t)b * TT) + t) * CC + h * HD + d] = (__bf16)(acc[mi][nt].f[r] * inv[r]);
      }
    }
  }
}

// ---------------- kernel 3: output projection (32x64 tile per wave) ----------------
__global__ __launch_bounds__(128) void proj_gemm(
    const __bf16* __restrict__ Ybf, const __bf16* __restrict__ W,
    const float* __restrict__ bias, float* __restrict__ out) {
  const int lane = threadIdx.x & 31;
  const int gw   = blockIdx.x * 4 + (threadIdx.x >> 5);
  const int NGRP = CC / 64;                       // 16
  const int mt = gw / NGRP, ng = gw % NGRP;       // mt over 128 tiles of 32 rows
  const int l16 = lane & 15, hi = (lane >> 4) & 1;

  size_t abase[2];
#pragma unroll
  for (int mi = 0; mi < 2; ++mi)
    abase[mi] = (size_t)(mt * 32 + mi * 16 + l16) * CC + (hi ? 8 : 0);
  const int nbase = ng * 64;
  size_t bbase[4];
#pragma unroll
  for (int nt = 0; nt < 4; ++nt)
    bbase[nt] = (size_t)(nbase + nt * 16 + l16) * CC + (hi ? 8 : 0);

  FragC acc[2][4];
#pragma unroll
  for (int mi = 0; mi < 2; ++mi)
#pragma unroll
    for (int nt = 0; nt < 4; ++nt)
#pragma unroll
      for (int r = 0; r < 8; ++r) acc[mi][nt].f[r] = 0.f;

  for (int k = 0; k < CC; k += 32) {
    FragAB a[2];
#pragma unroll
    for (int mi = 0; mi < 2; ++mi)
      load_frag(Ybf + abase[mi] + k, Ybf + abase[mi] + k + 16, a[mi]);
#pragma unroll
    for (int nt = 0; nt < 4; ++nt) {
      FragAB b;
      const __bf16* bp = W + bbase[nt] + k;
      load_frag(bp, bp + 16, b);
      acc[0][nt].v = wmma_bf16(a[0].v, b.v, acc[0][nt].v);
      acc[1][nt].v = wmma_bf16(a[1].v, b.v, acc[1][nt].v);
    }
  }

#pragma unroll
  for (int nt = 0; nt < 4; ++nt) {
    const int n = nbase + nt * 16 + l16;
    const float bn = bias[n];
#pragma unroll
    for (int mi = 0; mi < 2; ++mi)
#pragma unroll
      for (int r = 0; r < 8; ++r) {
        const int m = mt * 32 + mi * 16 + r + hi * 8;
        out[(size_t)m * CC + n] = acc[mi][nt].f[r] + bn;
      }
  }
}

// ---------------- launch ----------------
extern "C" void kernel_launch(void* const* d_in, const int* in_sizes, int n_in,
                              void* d_out, int out_size, void* d_ws, size_t ws_size,
                              hipStream_t stream) {
  const float* x      = (const float*)d_in[0];   // [B,T,C]
  const float* w_attn = (const float*)d_in[1];   // [3C,C]
  const float* b_attn = (const float*)d_in[2];   // [3C]
  const float* w_proj = (const float*)d_in[3];   // [C,C]
  const float* b_proj = (const float*)d_in[4];   // [C]
  float* out = (float*)d_out;

  const size_t nX  = (size_t)MROWS * CC;         // 4,194,304
  const size_t nWA = (size_t)N_QKV * CC;         // 3,145,728
  const size_t nWP = (size_t)CC * CC;            // 1,048,576
  const size_t nQ  = (size_t)BB * HH * TT * HD;  // 4,194,304

  char* p = (char*)d_ws;
  __bf16* xbf  = (__bf16*)p;               p += nX  * 2;
  __bf16* wabf = (__bf16*)p;               p += nWA * 2;
  __bf16* wpbf = (__bf16*)p;               p += nWP * 2;
  __bf16* Qb   = (__bf16*)p;               p += nQ  * 2;
  __bf16* Kb   = (__bf16*)p;               p += nQ  * 2;
  __bf16* Vt   = (__bf16*)p;               p += nQ  * 2;
  __bf16* Ybf  = (__bf16*)p;               p += nQ  * 2;

  cvt_f32_bf16<<<2048, 256, 0, stream>>>(x, xbf, (int)nX);
  cvt_f32_bf16<<<2048, 256, 0, stream>>>(w_attn, wabf, (int)nWA);
  cvt_f32_bf16<<<1024, 256, 0, stream>>>(w_proj, wpbf, (int)nWP);

  // 128 M-tiles(32) x 48 N-groups(64) = 6144 waves, 4 waves/block
  qkv_gemm<<<1536, 128, 0, stream>>>(xbf, wabf, b_attn, Qb, Kb, Vt);

  // B*H*(T/32) = 2048 waves
  attn<<<512, 128, 0, stream>>>(Qb, Kb, Vt, Ybf);

  // 128 x 16 = 2048 waves
  proj_gemm<<<512, 128, 0, stream>>>(Ybf, wpbf, b_proj, out);
}
